// relation_classifier_84610855731246
// MI455X (gfx1250) — matrix operand order repaired and checked by reading, not compiled, not run on hardware
//
#include <hip/hip_runtime.h>

// MI455X (gfx1250) 3-layer MLP, bf16 WMMA w/ f32 accumulation.
// Compute-bound (~210 GFLOP vs ~95MB HBM traffic) -> v_wmma_f32_16x16x32_bf16.
// Intermediate activations bf16 in workspace (sit in 192MB L2).
// Tiling: block 128x256x32, 8 waves, 64x64/wave = 4x4 frags -> 16 WMMA/K-step.
// Double-buffered LDS (one barrier per K-step); W-row index clamped (not
// predicated) so the K-loop has zero divergence.

typedef __bf16 bf16;
typedef bf16  bf16x4  __attribute__((ext_vector_type(4)));
typedef bf16  bf16x8  __attribute__((ext_vector_type(8)));
typedef bf16  bf16x16 __attribute__((ext_vector_type(16)));
typedef float floatx4 __attribute__((ext_vector_type(4)));
typedef float floatx8 __attribute__((ext_vector_type(8)));

#define BM 128
#define BN 256
#define BK 32
#define LDA_P 40   // padded row pitch (bf16 elems) for LDS A tile
#define LDB_P 40   // padded row pitch (bf16 elems) for LDS B tile

__device__ __forceinline__ bf16x16 cat16(bf16x8 lo, bf16x8 hi) {
  return __builtin_shufflevector(lo, hi, 0,1,2,3,4,5,6,7,8,9,10,11,12,13,14,15);
}

__device__ __forceinline__ bf16x8 cvt8(floatx4 a, floatx4 b) {
  bf16x8 r;
  r[0]=(bf16)a.x; r[1]=(bf16)a.y; r[2]=(bf16)a.z; r[3]=(bf16)a.w;
  r[4]=(bf16)b.x; r[5]=(bf16)b.y; r[6]=(bf16)b.z; r[7]=(bf16)b.w;
  return r;
}

// fp32 -> bf16 elementwise (weights). n4 = count/4.
__global__ void cvt_f32_to_bf16(const float* __restrict__ src,
                                bf16* __restrict__ dst, int n4) {
  int i = blockIdx.x * blockDim.x + threadIdx.x;
  if (i < n4) {
    floatx4 v = ((const floatx4*)src)[i];
    bf16x4 o;
    o.x = (bf16)v.x; o.y = (bf16)v.y; o.z = (bf16)v.z; o.w = (bf16)v.w;
    ((bf16x4*)dst)[i] = o;
  }
}

// C[m,n] = act( sum_k A[m,k]*W[n,k] + bias[n] )
// A: [M,K] row-major (float or bf16), W: [N,K] row-major bf16.
// M % 128 == 0, K % 32 == 0; N handled with clamped loads + guarded stores.
template <typename AT, typename OT, bool RELU>
__global__ __launch_bounds__(256)
void mlp_gemm(const AT* __restrict__ A, const bf16* __restrict__ W,
              const float* __restrict__ bias, OT* __restrict__ out,
              int M, int N, int K)
{
  __shared__ __align__(16) bf16 sA[2][BM * LDA_P];
  __shared__ __align__(16) bf16 sB[2][BN * LDB_P];

  const int tid   = threadIdx.x;
  const int lane  = tid & 31;
  const int half  = lane >> 4;   // lane group 0-15 vs 16-31
  const int lq    = lane & 15;
  const int wave  = tid >> 5;    // 8 waves
  const int waveM = wave & 1;    // 2 along M
  const int waveN = wave >> 1;   // 4 along N

  const int m0b = blockIdx.y * BM;
  const int n0b = blockIdx.x * BN;
  const int m0w = waveM * 64;
  const int n0w = waveN * 64;

  // staging assignments (256 threads)
  const int arow  = tid >> 1;          // A tile row 0..127
  const int akseg = (tid & 1) * 16;    // 16 K-elems per thread
  const int brow  = tid;               // B tile row 0..255
  // clamp instead of predicate: OOB lanes re-read row N-1 (finite values,
  // results discarded at store) -> no exec-mask divergence in the K-loop.
  const int browg = (n0b + brow < N) ? (n0b + brow) : (N - 1);

  const AT*   aptr = A + (size_t)(m0b + arow) * K + akseg;
  const bf16* bptr = W + (size_t)browg * K;

  floatx8 acc[4][4];
  #pragma unroll
  for (int i = 0; i < 4; ++i)
    #pragma unroll
    for (int j = 0; j < 4; ++j)
      acc[i][j] = (floatx8){};

  bf16x8 astg[2];
  bf16x8 bstg[4];

  auto gload = [&](int k0) {
    if constexpr (sizeof(AT) == 4) {
      const float* ap = (const float*)aptr + k0;
      floatx4 f0 = ((const floatx4*)ap)[0];
      floatx4 f1 = ((const floatx4*)ap)[1];
      floatx4 f2 = ((const floatx4*)ap)[2];
      floatx4 f3 = ((const floatx4*)ap)[3];
      astg[0] = cvt8(f0, f1);
      astg[1] = cvt8(f2, f3);
    } else {
      const bf16* ap = (const bf16*)aptr + k0;
      astg[0] = ((const bf16x8*)ap)[0];
      astg[1] = ((const bf16x8*)ap)[1];
    }
    const bf16* bp = bptr + k0;
    #pragma unroll
    for (int j = 0; j < 4; ++j)
      bstg[j] = ((const bf16x8*)bp)[j];
  };

  auto sstore = [&](int buf) {
    *(bf16x8*)&sA[buf][arow * LDA_P + akseg]     = astg[0];
    *(bf16x8*)&sA[buf][arow * LDA_P + akseg + 8] = astg[1];
    #pragma unroll
    for (int j = 0; j < 4; ++j)
      *(bf16x8*)&sB[buf][brow * LDB_P + j * 8] = bstg[j];
  };

  auto compute = [&](int buf) {
    // fragment loads: contiguous K = 16*half + e, same order for A and B
    bf16x16 af[4];
    #pragma unroll
    for (int mi = 0; mi < 4; ++mi) {
      const bf16* p = &sA[buf][(m0w + mi * 16 + lq) * LDA_P];
      bf16x8 lo = *(const bf16x8*)(p + 16 * half);      // K = 16*half + e
      bf16x8 hi = *(const bf16x8*)(p + 16 * half + 8);  // K = 16*half + 8 + e
      af[mi] = cat16(lo, hi);
    }
    bf16x16 bfv[4];
    #pragma unroll
    for (int ni = 0; ni < 4; ++ni) {
      const bf16* p = &sB[buf][(n0w + ni * 16 + lq) * LDB_P + 16 * half];
      bf16x8 lo = *(const bf16x8*)(p);                  // K = 16*half + e
      bf16x8 hi = *(const bf16x8*)(p + 8);
      bfv[ni] = cat16(lo, hi);
    }
    #pragma unroll
    for (int mi = 0; mi < 4; ++mi)
      #pragma unroll
      for (int ni = 0; ni < 4; ++ni)
        acc[mi][ni] = __builtin_amdgcn_wmma_f32_16x16x32_bf16(
            false, af[mi], false, bfv[ni], (short)0, acc[mi][ni], false, false);
  };

  // ---- double-buffered main loop: one barrier per K-step ----
  const int nk = K / BK;
  gload(0);
  sstore(0);
  __syncthreads();
  for (int kt = 0; kt + 1 < nk; ++kt) {
    gload((kt + 1) * BK);       // next tile in flight while we compute
    compute(kt & 1);
    sstore((kt + 1) & 1);       // other buffer: last read before the previous barrier
    __syncthreads();
  }
  compute((nk - 1) & 1);

  // ---- epilogue: bias + activation + store ----
  #pragma unroll
  for (int ni = 0; ni < 4; ++ni) {
    const int n = n0b + n0w + ni * 16 + lq;          // N = lane&15 (both halves)
    const float bv = (n < N) ? bias[n] : 0.0f;
    #pragma unroll
    for (int mi = 0; mi < 4; ++mi) {
      #pragma unroll
      for (int r = 0; r < 8; ++r) {
        const int m = m0b + m0w + mi * 16 + r + 8 * half;  // M = r + 8*half
        float v = acc[mi][ni][r] + bv;
        if (RELU) v = v > 0.0f ? v : 0.0f;
        if (n < N) out[(size_t)m * N + n] = (OT)v;
      }
    }
  }
}

extern "C" void kernel_launch(void* const* d_in, const int* in_sizes, int n_in,
                              void* d_out, int out_size, void* d_ws, size_t ws_size,
                              hipStream_t stream) {
  (void)in_sizes; (void)n_in; (void)out_size; (void)ws_size;
  // setup_inputs order: num_pairs, classifier_input, num_batches, W0,b0,W1,b1,W2,b2
  const float* X0 = (const float*)d_in[1];   // [16,1024,1024]
  const float* W0 = (const float*)d_in[3];   // [2048,1024]
  const float* b0 = (const float*)d_in[4];
  const float* W1 = (const float*)d_in[5];   // [2048,2048]
  const float* b1 = (const float*)d_in[6];
  const float* W2 = (const float*)d_in[7];   // [51,2048]
  const float* b2 = (const float*)d_in[8];
  float* out = (float*)d_out;                // [16,1024,51] fp32

  constexpr int M  = 16 * 1024;
  constexpr int D0 = 1024, D1 = 2048, D2 = 2048, D3 = 51;

  // workspace carve-up (bf16 elements); weight block ends 256B-aligned.
  bf16* W0b = (bf16*)d_ws;
  bf16* W1b = W0b + (size_t)D1 * D0;                 // 2,097,152
  bf16* W2b = W1b + (size_t)D2 * D1;                 // +4,194,304
  bf16* X1b = W2b + (size_t)D3 * D2;                 // +104,448 -> 12,791,808 B
  bf16* X2b = X1b + (size_t)M * D1;                  // +64 MB
  // total ws usage ~140.2 MB

  // 1) weights fp32 -> bf16
  {
    int n4;
    n4 = (D1 * D0) / 4; cvt_f32_to_bf16<<<(n4 + 255) / 256, 256, 0, stream>>>(W0, W0b, n4);
    n4 = (D2 * D1) / 4; cvt_f32_to_bf16<<<(n4 + 255) / 256, 256, 0, stream>>>(W1, W1b, n4);
    n4 = (D3 * D2) / 4; cvt_f32_to_bf16<<<(n4 + 255) / 256, 256, 0, stream>>>(W2, W2b, n4);
  }

  // 2) three fused GEMM layers
  dim3 blk(256);
  {
    dim3 grid((D1 + BN - 1) / BN, M / BM);  // 8 x 128
    mlp_gemm<float, bf16, true><<<grid, blk, 0, stream>>>(X0, W0b, b0, X1b, M, D1, D0);
  }
  {
    dim3 grid((D2 + BN - 1) / BN, M / BM);  // 8 x 128
    mlp_gemm<bf16, bf16, true><<<grid, blk, 0, stream>>>(X1b, W1b, b1, X2b, M, D2, D1);
  }
  {
    dim3 grid((D3 + BN - 1) / BN, M / BM);  // 1 x 128
    mlp_gemm<bf16, float, false><<<grid, blk, 0, stream>>>(X2b, W2b, b2, out, M, D3, D2);
  }
}
